// CustomMultiHeadAttention_60885456388601
// MI455X (gfx1250) — matrix operand order repaired and checked by reading, not compiled, not run on hardware
//
#include <hip/hip_runtime.h>

// ---------------- problem constants ----------------
#define EMBED   1024
#define NHEADS  16
#define HD      64
#define BATCH   2
#define SEQ     2048
#define MTOT    (BATCH * SEQ)
#define SCALE_F 0.125f                  // 1/sqrt(64)
#define ROWBLK  32                      // fused kernel: query rows per block
#define SPAD    2052                    // padded LDS row stride (floats): 2052 mod 64 = 4

typedef __attribute__((ext_vector_type(16))) _Float16 v16h;
typedef __attribute__((ext_vector_type(8)))  _Float16 v8h;
typedef __attribute__((ext_vector_type(8)))  float    v8f;

__device__ __forceinline__ v16h cat16(v8h a, v8h b) {
  return __builtin_shufflevector(a, b, 0,1,2,3,4,5,6,7,8,9,10,11,12,13,14,15);
}

enum { MODE_PROJ_QK = 0, MODE_PROJ_V = 1, MODE_OUT = 2 };

// ------------------------------------------------------------------
// Batched f32 -> f16 conversion (one-time; makes every GEMM operand
// loadable directly into WMMA fragments from global/L2).
// ------------------------------------------------------------------
struct CvtJobs {
  const float* src[8];
  _Float16*    dst[8];
  int          n[8];
};

__global__ __launch_bounds__(256)
void cvt_f16_kernel(CvtJobs jobs) {
  const int j = blockIdx.y;
  const int idx = (blockIdx.x * 256 + threadIdx.x) * 8;
  if (idx >= jobs.n[j]) return;
  const float* s = jobs.src[j] + idx;
  const float4 a = *(const float4*)s;
  const float4 b = *(const float4*)(s + 4);
  v8h o;
  o[0] = (_Float16)a.x; o[1] = (_Float16)a.y;
  o[2] = (_Float16)a.z; o[3] = (_Float16)a.w;
  o[4] = (_Float16)b.x; o[5] = (_Float16)b.y;
  o[6] = (_Float16)b.z; o[7] = (_Float16)b.w;
  *(v8h*)(jobs.dst[j] + idx) = o;
}

// ------------------------------------------------------------------
// Direct-f16 GEMM: C = A @ B^T + bias.  A [M,1024] f16, B [N,1024] f16,
// both K-contiguous -> WMMA fragments loaded straight from global (L2),
// no LDS, no barriers. Block 128x128 (8 waves), wave tile 32x64
// = 2x4 WMMA frags; per K-step: 12 x b128 loads -> 8 WMMAs.
// ------------------------------------------------------------------
template <int MODE>
__global__ __launch_bounds__(256)
void wmma_gemm16_kernel(const _Float16* __restrict__ Ag,
                        const _Float16* __restrict__ Bg,
                        const float* __restrict__ bias,
                        void* __restrict__ Cout)
{
  const int tid  = threadIdx.x;
  const int lane = tid & 31;
  const int wave = tid >> 5;
  const int wm   = wave >> 1;               // 0..3
  const int wn   = wave & 1;                // 0..1
  const int ll   = lane & 15;
  const int hi   = lane >> 4;

  const int tileM = blockIdx.y * 128 + wm * 32;
  const int tileN = blockIdx.x * 128 + wn * 64;

  v8f acc[2][4] = {};

  for (int k0 = 0; k0 < EMBED; k0 += 32) {
    v16h af[2], bf[4];
#pragma unroll
    for (int t = 0; t < 2; ++t) {
      const _Float16* ap = Ag + (size_t)(tileM + t * 16 + ll) * EMBED + k0 + hi * 8;
      af[t] = cat16(*(const v8h*)ap, *(const v8h*)(ap + 16));
    }
#pragma unroll
    for (int t = 0; t < 4; ++t) {
      const _Float16* bp = Bg + (size_t)(tileN + t * 16 + ll) * EMBED + k0 + hi * 8;
      bf[t] = cat16(*(const v8h*)bp, *(const v8h*)(bp + 16));
    }
#pragma unroll
    for (int tm = 0; tm < 2; ++tm)
#pragma unroll
      for (int tn = 0; tn < 4; ++tn)
        acc[tm][tn] = __builtin_amdgcn_wmma_f32_16x16x32_f16(
            false, af[tm], false, bf[tn], (short)0, acc[tm][tn], false, false);
  }

#pragma unroll
  for (int tm = 0; tm < 2; ++tm) {
#pragma unroll
    for (int tn = 0; tn < 4; ++tn) {
      const int nloc = tileN + tn * 16 + ll;
      const float bv = bias[nloc];
#pragma unroll
      for (int r = 0; r < 8; ++r) {
        const int mloc = tileM + tm * 16 + hi * 8 + r;
        const float val = acc[tm][tn][r] + bv;
        if (MODE == MODE_PROJ_QK) {            // -> f16 [B,H,S,hd]
          const int b = mloc >> 11, s = mloc & (SEQ - 1);
          const int h = nloc >> 6,  d = nloc & (HD - 1);
          ((_Float16*)Cout)[(((size_t)(b * NHEADS + h) * SEQ + s) * HD + d)] =
              (_Float16)val;
        } else if (MODE == MODE_PROJ_V) {      // -> f16 [B,H,hd,S]
          const int b = mloc >> 11, s = mloc & (SEQ - 1);
          const int h = nloc >> 6,  d = nloc & (HD - 1);
          ((_Float16*)Cout)[(((size_t)(b * NHEADS + h) * HD + d) * SEQ + s)] =
              (_Float16)val;
        } else {                               // MODE_OUT -> f32 [M,E]
          ((float*)Cout)[(size_t)mloc * EMBED + nloc] = val;
        }
      }
    }
  }
}

// ------------------------------------------------------------------
// Fused attention: per (head z, 32-row block):
//   phase 1: S = scale * Q K^T  -> LDS (32 x 2048 f32, padded rows)
//   phase 2: masked softmax in LDS; weights written to d_out once
//   phase 3: O = P V via WMMA, P from LDS -> Ctx (f16)
// LDS ~264 KB (CDNA5: 320 KB/WGP). Dynamic shared memory.
// ------------------------------------------------------------------
__global__ __launch_bounds__(256)
void fused_attn_kernel(const _Float16* __restrict__ Qh,   // [B,H,S,hd]
                       const _Float16* __restrict__ Kh,   // [B,H,S,hd]
                       const _Float16* __restrict__ Vth,  // [B,H,hd,S]
                       const unsigned char* __restrict__ mask, // [B,1,S,S]
                       float* __restrict__ attnOut,       // [B,H,S,S]
                       _Float16* __restrict__ Ctx)        // [B,S,E] f16
{
  extern __shared__ char smem_raw[];
  float* shS  = (float*)smem_raw;          // [ROWBLK][SPAD]
  float* red  = shS + ROWBLK * SPAD;       // [8][32]
  float* rmax = red + 8 * 32;              // [32]
  float* rinv = rmax + 32;                 // [32]

  const int tid  = threadIdx.x;
  const int lane = tid & 31;
  const int wave = tid >> 5;
  const int ll   = lane & 15;
  const int hi   = lane >> 4;
  const int z    = blockIdx.z;             // b*H + h
  const int rb   = blockIdx.x;             // 32-row block index
  const int b    = z >> 4;
  const int h    = z & (NHEADS - 1);

  const _Float16* Qhead = Qh  + (size_t)z * SEQ * HD;
  const _Float16* Khead = Kh  + (size_t)z * SEQ * HD;
  const _Float16* Vhead = Vth + (size_t)z * HD * SEQ;
  float* Ahead = attnOut + (size_t)z * SEQ * SEQ + (size_t)rb * ROWBLK * SEQ;

  // ---- Q fragments (A-matrices), register resident for all of phase 1
  v16h aQ[2][2];
#pragma unroll
  for (int rt = 0; rt < 2; ++rt)
#pragma unroll
    for (int kk = 0; kk < 2; ++kk) {
      const _Float16* qp =
          Qhead + (size_t)(rb * ROWBLK + rt * 16 + ll) * HD + kk * 32 + hi * 8;
      aQ[rt][kk] = cat16(*(const v8h*)qp, *(const v8h*)(qp + 16));
    }

  // ---- phase 1: scores -> LDS, wave w owns column tiles w*16 .. w*16+15
#pragma unroll 2
  for (int ct = 0; ct < 16; ++ct) {
    const int colt = wave * 16 + ct;       // global 16-col tile, 0..127
    v8f s0 = {}, s1 = {};
#pragma unroll
    for (int kk = 0; kk < 2; ++kk) {
      const _Float16* kp =
          Khead + (size_t)(colt * 16 + ll) * HD + kk * 32 + hi * 8;
      v16h bfv = cat16(*(const v8h*)kp, *(const v8h*)(kp + 16));
      s0 = __builtin_amdgcn_wmma_f32_16x16x32_f16(false, aQ[0][kk], false, bfv,
                                                  (short)0, s0, false, false);
      s1 = __builtin_amdgcn_wmma_f32_16x16x32_f16(false, aQ[1][kk], false, bfv,
                                                  (short)0, s1, false, false);
    }
    const int col = colt * 16 + ll;
#pragma unroll
    for (int r = 0; r < 8; ++r) {
      shS[(hi * 8 + r) * SPAD + col]      = s0[r] * SCALE_F;
      shS[(16 + hi * 8 + r) * SPAD + col] = s1[r] * SCALE_F;
    }
  }
  __syncthreads();

  // ---- phase 2a: mask + row max (thread t: row t&31, 256-col chunk t>>5)
  const int srow  = tid & 31;
  const int chunk = tid >> 5;
  const int sq    = rb * ROWBLK + srow;
  const unsigned char* mp = mask + ((size_t)b * SEQ + sq) * SEQ + chunk * 256;
  float* sp = shS + srow * SPAD + chunk * 256;

  float mx = -INFINITY;
#pragma unroll 4
  for (int i = 0; i < 64; ++i) {
    float4 v = *(float4*)(sp + i * 4);
    const unsigned int mm = *(const unsigned int*)(mp + i * 4);
    if (!(mm & 0x000000ffu)) v.x = -INFINITY;
    if (!(mm & 0x0000ff00u)) v.y = -INFINITY;
    if (!(mm & 0x00ff0000u)) v.z = -INFINITY;
    if (!(mm & 0xff000000u)) v.w = -INFINITY;
    *(float4*)(sp + i * 4) = v;
    mx = fmaxf(mx, fmaxf(fmaxf(v.x, v.y), fmaxf(v.z, v.w)));
  }
  red[chunk * 32 + srow] = mx;
  __syncthreads();
  if (tid < 32) {
    float m2 = red[tid];
#pragma unroll
    for (int c = 1; c < 8; ++c) m2 = fmaxf(m2, red[c * 32 + tid]);
    rmax[tid] = m2;
  }
  __syncthreads();

  // ---- phase 2b: exp + row sum
  const float rowm = rmax[srow];
  float sum = 0.0f;
#pragma unroll 4
  for (int i = 0; i < 64; ++i) {
    float4 v = *(float4*)(sp + i * 4);
    v.x = __expf(v.x - rowm); v.y = __expf(v.y - rowm);
    v.z = __expf(v.z - rowm); v.w = __expf(v.w - rowm);
    *(float4*)(sp + i * 4) = v;
    sum += (v.x + v.y) + (v.z + v.w);
  }
  red[chunk * 32 + srow] = sum;
  __syncthreads();
  if (tid < 32) {
    float s2 = 0.0f;
#pragma unroll
    for (int c = 0; c < 8; ++c) s2 += red[c * 32 + tid];
    rinv[tid] = 1.0f / s2;
  }
  __syncthreads();

  // ---- phase 2c: normalize in LDS + single coalesced weight write
#pragma unroll 4
  for (int i = 0; i < 64; ++i) {
    const int off = (i * 256 + tid) * 4;      // 0 .. 65535, coalesced
    const int row = off >> 11;
    const int col = off & (SEQ - 1);
    const float inv = rinv[row];
    float4 v = *(float4*)(shS + row * SPAD + col);
    v.x *= inv; v.y *= inv; v.z *= inv; v.w *= inv;
    *(float4*)(shS + row * SPAD + col) = v;
    *(float4*)(Ahead + off) = v;
  }
  __syncthreads();

  // ---- phase 3: O = P V. M=32,N=64,K=2048. One 16x16 tile per wave.
  const int rt = wave >> 2;                  // row tile 0..1
  const int nt = wave & 3;                   // col tile 0..3
  v8f acc = {};
  const float* arow = shS + (rt * 16 + ll) * SPAD;
  const _Float16* vrow = Vhead + (size_t)(nt * 16 + ll) * SEQ;

  for (int k0 = 0; k0 < SEQ; k0 += 32) {
    const float* ap = arow + k0 + hi * 8;
    v16h af;
#pragma unroll
    for (int e = 0; e < 8; ++e) af[e]     = (_Float16)ap[e];
#pragma unroll
    for (int e = 0; e < 8; ++e) af[e + 8] = (_Float16)ap[16 + e];
    const _Float16* vp = vrow + k0 + hi * 8;
    v16h bfv = cat16(*(const v8h*)vp, *(const v8h*)(vp + 16));
    acc = __builtin_amdgcn_wmma_f32_16x16x32_f16(false, af, false, bfv,
                                                 (short)0, acc, false, false);
  }

  const int ncol = h * HD + nt * 16 + ll;
#pragma unroll
  for (int r = 0; r < 8; ++r) {
    const int mm = rb * ROWBLK + rt * 16 + hi * 8 + r;
    Ctx[(size_t)(b * SEQ + mm) * EMBED + ncol] = (_Float16)acc[r];
  }
}

extern "C" void kernel_launch(void* const* d_in, const int* in_sizes, int n_in,
                              void* d_out, int out_size, void* d_ws, size_t ws_size,
                              hipStream_t stream) {
  const float* x_q = (const float*)d_in[0];
  const float* x_k = (const float*)d_in[1];
  const float* x_v = (const float*)d_in[2];
  const unsigned char* mask = (const unsigned char*)d_in[3];
  const float* Wq = (const float*)d_in[4];  const float* bq = (const float*)d_in[5];
  const float* Wk = (const float*)d_in[6];  const float* bk = (const float*)d_in[7];
  const float* Wv = (const float*)d_in[8];  const float* bv = (const float*)d_in[9];
  const float* Wo = (const float*)d_in[10]; const float* bo = (const float*)d_in[11];

  float* out  = (float*)d_out;                              // [B,S,E]
  float* attn = out + (size_t)BATCH * SEQ * EMBED;          // [B,H,S,S]

  // ---- f16 workspace layout (~67 MB total)
  const size_t NXE = (size_t)MTOT * EMBED;                  // 4.19M
  const size_t NWW = (size_t)EMBED * EMBED;                 // 1.05M
  _Float16* x16q = (_Float16*)d_ws;
  _Float16* x16k = x16q + NXE;
  _Float16* x16v = x16k + NXE;
  _Float16* W16q = x16v + NXE;
  _Float16* W16k = W16q + NWW;
  _Float16* W16v = W16k + NWW;
  _Float16* W16o = W16v + NWW;
  _Float16* Qh   = W16o + NWW;                              // [B,H,S,hd]
  _Float16* Kh   = Qh + NXE;
  _Float16* Vth  = Kh + NXE;                                // [B,H,hd,S]
  _Float16* Ctx  = Vth + NXE;                               // [B,S,E] f16

  const dim3 blk(256);

  // 0: one-time f32->f16 conversion of all GEMM operands (7 jobs)
  CvtJobs jobs = {};
  jobs.src[0] = x_q; jobs.dst[0] = x16q; jobs.n[0] = (int)NXE;
  jobs.src[1] = x_k; jobs.dst[1] = x16k; jobs.n[1] = (int)NXE;
  jobs.src[2] = x_v; jobs.dst[2] = x16v; jobs.n[2] = (int)NXE;
  jobs.src[3] = Wq;  jobs.dst[3] = W16q; jobs.n[3] = (int)NWW;
  jobs.src[4] = Wk;  jobs.dst[4] = W16k; jobs.n[4] = (int)NWW;
  jobs.src[5] = Wv;  jobs.dst[5] = W16v; jobs.n[5] = (int)NWW;
  jobs.src[6] = Wo;  jobs.dst[6] = W16o; jobs.n[6] = (int)NWW;
  jobs.src[7] = Wo;  jobs.dst[7] = W16o; jobs.n[7] = 0;     // unused slot
  cvt_f16_kernel<<<dim3((unsigned)(NXE / (256 * 8)), 7), blk, 0, stream>>>(jobs);

  // 1-3: projections (M=4096, N=1024, K=1024), direct-f16, no LDS
  const dim3 gproj(EMBED / 128, MTOT / 128, 1);
  wmma_gemm16_kernel<MODE_PROJ_QK><<<gproj, blk, 0, stream>>>(x16q, W16q, bq, Qh);
  wmma_gemm16_kernel<MODE_PROJ_QK><<<gproj, blk, 0, stream>>>(x16k, W16k, bk, Kh);
  wmma_gemm16_kernel<MODE_PROJ_V ><<<gproj, blk, 0, stream>>>(x16v, W16v, bv, Vth);

  // 4: fused scores + softmax + P*V; weights hit HBM exactly once
  const size_t ldsBytes = (size_t)(ROWBLK * SPAD + 8 * 32 + 64) * sizeof(float);
  const dim3 gfused(SEQ / ROWBLK, 1, BATCH * NHEADS);
  fused_attn_kernel<<<gfused, blk, ldsBytes, stream>>>(Qh, Kh, Vth, mask, attn, Ctx);

  // 5: output projection (M=4096, N=1024, K=1024), direct-f16 A=Ctx
  const dim3 gout(EMBED / 128, MTOT / 128, 1);
  wmma_gemm16_kernel<MODE_OUT><<<gout, blk, 0, stream>>>(Ctx, W16o, bo, out);
}
